// ProtoWRN_32195074850856
// MI455X (gfx1250) — matrix operand order repaired and checked by reading, not compiled, not run on hardware
//
#include <hip/hip_runtime.h>
#include <hip/hip_bf16.h>

// ---------------------------------------------------------------------------
// ProtoWRN loss for MI455X (gfx1250, wave32, WMMA + async-to-LDS).
// cross = z @ protos^T via split-bf16 (hi/lo) 3x v_wmma_f32_16x16x32_bf16.
// dists = ||z||^2 + ||p||^2 - 2*cross ; scores = 1/(dists+eps);
// loss = mean CE of log_softmax(-dists) at targets.
//
// Fast path (ws large enough): prep kernels pre-split z/protos into bf16
// hi/lo arrays in ws; GEMM stages tiles with GLOBAL_LOAD_ASYNC_TO_LDS_B128
// (ASYNCcnt, double-buffered LDS) and runs pure WMMA.
// Fallback path: convert-in-kernel GEMM (round-1 style).
// ---------------------------------------------------------------------------

#define BATCH 16384
#define NCLS  1000
#define DIM   1024
#define EPSF  1e-10f

#define TM 128   // block tile M (batch rows)
#define TN 128   // block tile N (classes)
#define KC 32    // K chunk (bf16 WMMA K)
#define LP 40    // LDS pitch in bf16 elems (32 data + 8 pad: 20-bank row stride
                 // -> the 16 rows of a half-wave b128 load cover all 64 banks)

#define TILE_USH (TM * LP)        // 5120 ushorts (10240 B) per tile
#define BUF_USH  (4 * TILE_USH)   // zh,zl,ph,pl tiles: 40960 B per buffer

typedef __attribute__((ext_vector_type(16))) __bf16          v16bf;
typedef __attribute__((ext_vector_type(8)))  float           v8f;
typedef __attribute__((ext_vector_type(8)))  unsigned short  u16x8;
typedef __attribute__((ext_vector_type(4)))  unsigned short  u16x4;

static __device__ __forceinline__ unsigned short bf16_rne(float x) {
  unsigned int u = __float_as_uint(x);
  u += 0x7FFFu + ((u >> 16) & 1u);           // round-to-nearest-even
  return (unsigned short)(u >> 16);
}
static __device__ __forceinline__ float bf16_f32(unsigned short h) {
  return __uint_as_float(((unsigned int)h) << 16);
}
static __device__ __forceinline__ void cvt4(float4 a, u16x4* h, u16x4* l) {
  float v[4] = {a.x, a.y, a.z, a.w};
#pragma unroll
  for (int j = 0; j < 4; ++j) {
    unsigned short hb = bf16_rne(v[j]);
    (*h)[j] = hb;
    (*l)[j] = bf16_rne(v[j] - bf16_f32(hb));
  }
}

union V16 { v16bf bf; u16x8 h[2]; };

// CDNA5 async copy: global -> LDS, per-lane 16B, tracked by ASYNCcnt.
static __device__ __forceinline__ void async_ld_b128(unsigned lds_off,
                                                     const void* gaddr) {
  asm volatile("global_load_async_to_lds_b128 %0, %1, off"
               :: "v"(lds_off), "v"(gaddr) : "memory");
}
static __device__ __forceinline__ void wait_async0() {
  asm volatile("s_wait_asynccnt 0x0" ::: "memory");
}

// ---------------- prototype prep: p_sq (+ optional bf16 hi/lo pack) --------
__global__ __launch_bounds__(256) void prep_p(const float* __restrict__ pr,
                                              float* __restrict__ psq,
                                              float* __restrict__ loss,
                                              unsigned short* __restrict__ ph,
                                              unsigned short* __restrict__ pl,
                                              int do_pack) {
  __shared__ float red[8];
  const int c = blockIdx.x;          // 0..1023 (rows >= 1000 are zero padding)
  const int tid = threadIdx.x;       // 256 threads, one float4 each (1024 elems)
  float4 v = make_float4(0.f, 0.f, 0.f, 0.f);
  if (c < NCLS) v = ((const float4*)(pr + (size_t)c * DIM))[tid];
  if (do_pack) {
    u16x4 h4, l4;
    cvt4(v, &h4, &l4);
    *(u16x4*)(ph + (size_t)c * DIM + tid * 4) = h4;
    *(u16x4*)(pl + (size_t)c * DIM + tid * 4) = l4;
  }
  float s = v.x * v.x + v.y * v.y + v.z * v.z + v.w * v.w;
#pragma unroll
  for (int o = 16; o; o >>= 1) s += __shfl_xor(s, o, 32);
  if ((tid & 31) == 0) red[tid >> 5] = s;
  __syncthreads();
  if (tid == 0) {
    float t = 0.f;
#pragma unroll
    for (int w = 0; w < 8; ++w) t += red[w];
    if (c < NCLS) psq[c] = t;
    if (c == 0) *loss = 0.f;         // zero loss accumulator early in stream
  }
}

// --------- z prep: passthrough copy + z_sq (+ optional hi/lo pack) ---------
__global__ __launch_bounds__(256) void prep_z(const float* __restrict__ z,
                                              float* __restrict__ zcopy,
                                              float* __restrict__ zsq,
                                              unsigned short* __restrict__ zh,
                                              unsigned short* __restrict__ zl,
                                              int do_pack) {
  __shared__ float red[8];
  const int b = blockIdx.x;          // 0..16383
  const int tid = threadIdx.x;
  float4 v = ((const float4*)(z + (size_t)b * DIM))[tid];
  // zcopy base is d_out+1 (4B aligned only) -> scalar stores
  float* o = zcopy + (size_t)b * DIM + tid * 4;
  o[0] = v.x; o[1] = v.y; o[2] = v.z; o[3] = v.w;
  if (do_pack) {
    u16x4 h4, l4;
    cvt4(v, &h4, &l4);
    *(u16x4*)(zh + (size_t)b * DIM + tid * 4) = h4;
    *(u16x4*)(zl + (size_t)b * DIM + tid * 4) = l4;
  }
  float s = v.x * v.x + v.y * v.y + v.z * v.z + v.w * v.w;
#pragma unroll
  for (int o2 = 16; o2; o2 >>= 1) s += __shfl_xor(s, o2, 32);
  if ((tid & 31) == 0) red[tid >> 5] = s;
  __syncthreads();
  if (tid == 0) {
    float t = 0.f;
#pragma unroll
    for (int w = 0; w < 8; ++w) t += red[w];
    zsq[b] = t;
  }
}

// ----------------- shared epilogue: dists -> scores ------------------------
static __device__ __forceinline__ void scores_epilogue(
    const v8f acc[2][4], const float* __restrict__ zsq,
    const float* __restrict__ psq, float* __restrict__ scores,
    int bM, int bN, int wm, int wn, int l16, int lhi) {
#pragma unroll
  for (int tm = 0; tm < 2; ++tm) {
    const int rowb = bM + wm * 32 + tm * 16 + lhi * 8;
    float zs[8];
#pragma unroll
    for (int r = 0; r < 8; ++r) zs[r] = zsq[rowb + r];
#pragma unroll
    for (int tn = 0; tn < 4; ++tn) {
      const int col = bN + wn * 64 + tn * 16 + l16;
      if (col < NCLS) {
        const float ps = psq[col];
#pragma unroll
        for (int r = 0; r < 8; ++r) {
          float dist = zs[r] + ps - 2.0f * acc[tm][tn][r];
          scores[(size_t)(rowb + r) * NCLS + col] = 1.0f / (dist + EPSF);
        }
      }
    }
  }
}

// ---------- fast GEMM: async-to-LDS staging of pre-packed bf16 -------------
__global__ __launch_bounds__(256) void proto_gemm_async(
    const unsigned short* __restrict__ zh, const unsigned short* __restrict__ zl,
    const unsigned short* __restrict__ ph, const unsigned short* __restrict__ pl,
    const float* __restrict__ zsq, const float* __restrict__ psq,
    float* __restrict__ scores) {
  extern __shared__ __align__(16) unsigned short smem[];  // 2 * 40960 B

  const int bN   = blockIdx.x * TN;   // grid.x = 8 (N fastest: z panel L2-resident)
  const int bM   = blockIdx.y * TM;   // grid.y = 128
  const int tid  = threadIdx.x;
  const int lane = tid & 31;
  const int wave = tid >> 5;          // 8 waves
  const int l16  = lane & 15;
  const int lhi  = lane >> 4;
  const int wm   = wave & 3;          // 4 waves along M
  const int wn   = wave >> 2;         // 2 waves along N

  // staging role: wave -> (tile, row half); lane -> (row-in-8, 16B chunk)
  const int st  = wave >> 1;          // tile 0..3 : zh,zl,ph,pl
  const int sh  = wave & 1;           // row half 0/1 (64 rows each)
  const int srl = lane >> 2;          // 0..7
  const int sj  = lane & 3;           // 16B chunk in 64B row
  const unsigned short* srcs[4] = {zh, zl, ph, pl};
  const unsigned short* ssrc = srcs[st] + (size_t)((st < 2) ? bM : bN) * DIM;
  const unsigned smem_base = (unsigned)(size_t)smem;  // LDS byte offset

  v8f acc[2][4];
#pragma unroll
  for (int i = 0; i < 2; ++i)
#pragma unroll
    for (int j = 0; j < 4; ++j) acc[i][j] = {};

  // issue one K-chunk of async copies into LDS buffer `buf`
  auto issue = [&](int buf, int k0) {
    const unsigned ldsb =
        smem_base + (unsigned)(buf * BUF_USH + st * TILE_USH) * 2u;
#pragma unroll
    for (int i = 0; i < 8; ++i) {
      const int r = sh * 64 + i * 8 + srl;                 // 0..127
      const unsigned loff = ldsb + (unsigned)(r * LP + sj * 8) * 2u;
      async_ld_b128(loff, ssrc + (size_t)r * DIM + k0 + sj * 8);
    }
  };

  issue(0, 0);
  int buf = 0;
  for (int k0 = 0; k0 < DIM; k0 += KC, buf ^= 1) {
    wait_async0();        // this wave's copies done
    __syncthreads();      // everyone's copies done; prev reads of other buf done
    if (k0 + KC < DIM) issue(buf ^ 1, k0 + KC);  // prefetch next chunk

    const unsigned short* sb  = smem + buf * BUF_USH;
    const unsigned short* sAh = sb;
    const unsigned short* sAl = sb + TILE_USH;
    const unsigned short* sBh = sb + 2 * TILE_USH;
    const unsigned short* sBl = sb + 3 * TILE_USH;

    // A 16x32 bf16: lane<16 holds K[0..7],K[16..23]; lane>=16 K[8..15],K[24..31]
    v16bf Ah[2], Al[2];
#pragma unroll
    for (int tm = 0; tm < 2; ++tm) {
      const int r = wm * 32 + tm * 16 + l16;
      V16 t;
      t.h[0] = *(const u16x8*)(sAh + r * LP + lhi * 8);
      t.h[1] = *(const u16x8*)(sAh + r * LP + 16 + lhi * 8);
      Ah[tm] = t.bf;
      t.h[0] = *(const u16x8*)(sAl + r * LP + lhi * 8);
      t.h[1] = *(const u16x8*)(sAl + r * LP + 16 + lhi * 8);
      Al[tm] = t.bf;
    }
    // B 32x16 bf16: lane holds col l16, K[lhi*8..+7], K[16+lhi*8..+7]
    v16bf Bh[4], Bl[4];
#pragma unroll
    for (int tn = 0; tn < 4; ++tn) {
      const int r = wn * 64 + tn * 16 + l16;
      V16 t;
      t.h[0] = *(const u16x8*)(sBh + r * LP + lhi * 8);
      t.h[1] = *(const u16x8*)(sBh + r * LP + 16 + lhi * 8);
      Bh[tn] = t.bf;
      t.h[0] = *(const u16x8*)(sBl + r * LP + lhi * 8);
      t.h[1] = *(const u16x8*)(sBl + r * LP + 16 + lhi * 8);
      Bl[tn] = t.bf;
    }

    // split-bf16 FMA: acc += hi*hi + hi*lo + lo*hi (near-f32 accuracy)
#pragma unroll
    for (int tm = 0; tm < 2; ++tm)
#pragma unroll
      for (int tn = 0; tn < 4; ++tn) {
        acc[tm][tn] = __builtin_amdgcn_wmma_f32_16x16x32_bf16(
            false, Ah[tm], false, Bh[tn], (short)0, acc[tm][tn], false, false);
        acc[tm][tn] = __builtin_amdgcn_wmma_f32_16x16x32_bf16(
            false, Ah[tm], false, Bl[tn], (short)0, acc[tm][tn], false, false);
        acc[tm][tn] = __builtin_amdgcn_wmma_f32_16x16x32_bf16(
            false, Al[tm], false, Bh[tn], (short)0, acc[tm][tn], false, false);
      }
  }

  scores_epilogue(acc, zsq, psq, scores, bM, bN, wm, wn, l16, lhi);
}

// ---------- fallback GEMM: convert-in-kernel (small ws) --------------------
__global__ __launch_bounds__(256) void proto_gemm_cvt(
    const float* __restrict__ z, const float* __restrict__ pr,
    const float* __restrict__ zsq, const float* __restrict__ psq,
    float* __restrict__ scores) {
  __shared__ __align__(16) unsigned short sAh[TM * LP];
  __shared__ __align__(16) unsigned short sAl[TM * LP];
  __shared__ __align__(16) unsigned short sBh[TN * LP];
  __shared__ __align__(16) unsigned short sBl[TN * LP];

  const int bN   = blockIdx.x * TN;
  const int bM   = blockIdx.y * TM;
  const int tid  = threadIdx.x;
  const int lane = tid & 31;
  const int wave = tid >> 5;
  const int l16  = lane & 15;
  const int lhi  = lane >> 4;
  const int wm   = wave & 3;
  const int wn   = wave >> 2;

  v8f acc[2][4];
#pragma unroll
  for (int i = 0; i < 2; ++i)
#pragma unroll
    for (int j = 0; j < 4; ++j) acc[i][j] = {};

  for (int k0 = 0; k0 < DIM; k0 += KC) {
    __syncthreads();
#pragma unroll
    for (int i = 0; i < 4; ++i) {
      const int p   = tid + i * 256;
      const int row = p >> 3;
      const int c4  = (p & 7) << 2;
      float4 a = *(const float4*)(z + (size_t)(bM + row) * DIM + k0 + c4);
      u16x4 ah, al;
      cvt4(a, &ah, &al);
      *(u16x4*)(sAh + row * LP + c4) = ah;
      *(u16x4*)(sAl + row * LP + c4) = al;
      const int pc = bN + row;
      float4 b = make_float4(0.f, 0.f, 0.f, 0.f);
      if (pc < NCLS) b = *(const float4*)(pr + (size_t)pc * DIM + k0 + c4);
      u16x4 bh, bl;
      cvt4(b, &bh, &bl);
      *(u16x4*)(sBh + row * LP + c4) = bh;
      *(u16x4*)(sBl + row * LP + c4) = bl;
    }
    __syncthreads();

    v16bf Ah[2], Al[2];
#pragma unroll
    for (int tm = 0; tm < 2; ++tm) {
      const int r = wm * 32 + tm * 16 + l16;
      V16 t;
      t.h[0] = *(const u16x8*)(sAh + r * LP + lhi * 8);
      t.h[1] = *(const u16x8*)(sAh + r * LP + 16 + lhi * 8);
      Ah[tm] = t.bf;
      t.h[0] = *(const u16x8*)(sAl + r * LP + lhi * 8);
      t.h[1] = *(const u16x8*)(sAl + r * LP + 16 + lhi * 8);
      Al[tm] = t.bf;
    }
    v16bf Bh[4], Bl[4];
#pragma unroll
    for (int tn = 0; tn < 4; ++tn) {
      const int r = wn * 64 + tn * 16 + l16;
      V16 t;
      t.h[0] = *(const u16x8*)(sBh + r * LP + lhi * 8);
      t.h[1] = *(const u16x8*)(sBh + r * LP + 16 + lhi * 8);
      Bh[tn] = t.bf;
      t.h[0] = *(const u16x8*)(sBl + r * LP + lhi * 8);
      t.h[1] = *(const u16x8*)(sBl + r * LP + 16 + lhi * 8);
      Bl[tn] = t.bf;
    }
#pragma unroll
    for (int tm = 0; tm < 2; ++tm)
#pragma unroll
      for (int tn = 0; tn < 4; ++tn) {
        acc[tm][tn] = __builtin_amdgcn_wmma_f32_16x16x32_bf16(
            false, Ah[tm], false, Bh[tn], (short)0, acc[tm][tn], false, false);
        acc[tm][tn] = __builtin_amdgcn_wmma_f32_16x16x32_bf16(
            false, Ah[tm], false, Bl[tn], (short)0, acc[tm][tn], false, false);
        acc[tm][tn] = __builtin_amdgcn_wmma_f32_16x16x32_bf16(
            false, Al[tm], false, Bh[tn], (short)0, acc[tm][tn], false, false);
      }
  }

  scores_epilogue(acc, zsq, psq, scores, bM, bN, wm, wn, l16, lhi);
}

// ------------------- per-row log-softmax CE + mean -------------------------
__global__ __launch_bounds__(256) void proto_loss(
    const float* __restrict__ scores, const int* __restrict__ targets,
    float* __restrict__ loss) {
  __shared__ float red[8];
  const int b = blockIdx.x;
  const int tid = threadIdx.x;
  const float* row = scores + (size_t)b * NCLS;

  // logit = -dist = eps - 1/score (exact inverse of what we stored)
  float lg[4];
  float m = -3.4e38f;
#pragma unroll
  for (int i = 0; i < 4; ++i) {
    const int c = tid + i * 256;
    lg[i] = (c < NCLS) ? (EPSF - 1.0f / row[c]) : -3.4e38f;
    m = fmaxf(m, lg[i]);
  }
#pragma unroll
  for (int o = 16; o; o >>= 1) m = fmaxf(m, __shfl_xor(m, o, 32));
  if ((tid & 31) == 0) red[tid >> 5] = m;
  __syncthreads();
  if (tid < 32) {
    float r = (tid < 8) ? red[tid] : -3.4e38f;
#pragma unroll
    for (int o = 4; o; o >>= 1) r = fmaxf(r, __shfl_xor(r, o, 32));
    if (tid == 0) red[0] = r;
  }
  __syncthreads();
  m = red[0];
  __syncthreads();

  float s = 0.f;
#pragma unroll
  for (int i = 0; i < 4; ++i) s += __expf(lg[i] - m);  // pads give exp(-inf)=0
#pragma unroll
  for (int o = 16; o; o >>= 1) s += __shfl_xor(s, o, 32);
  if ((tid & 31) == 0) red[tid >> 5] = s;
  __syncthreads();
  if (tid == 0) {
    float tot = 0.f;
#pragma unroll
    for (int w = 0; w < 8; ++w) tot += red[w];
    const int t = targets[b];
    const float lt = EPSF - 1.0f / row[t];
    atomicAdd(loss, ((m + __logf(tot)) - lt) * (1.0f / (float)BATCH));
  }
}

// ---------------------------------------------------------------------------
extern "C" void kernel_launch(void* const* d_in, const int* in_sizes, int n_in,
                              void* d_out, int out_size, void* d_ws, size_t ws_size,
                              hipStream_t stream) {
  const float* z  = (const float*)d_in[0];   // [16384,1024] f32
  const float* pr = (const float*)d_in[1];   // [1000,1024] f32
  const int*   tg = (const int*)d_in[2];     // [16384] i32

  float* out    = (float*)d_out;
  float* loss   = out;                               // [1]
  float* zcopy  = out + 1;                           // [16384*1024]
  float* scores = out + 1 + (size_t)BATCH * DIM;     // [16384*1000]

  // workspace layout
  char* ws = (char*)d_ws;
  size_t off = 0;
  float* psq = (float*)(ws + off);            off += 1024 * sizeof(float);
  float* zsq = (float*)(ws + off);            off += (size_t)BATCH * sizeof(float);
  unsigned short* ph = (unsigned short*)(ws + off); off += (size_t)1024 * DIM * 2;
  unsigned short* pl = (unsigned short*)(ws + off); off += (size_t)1024 * DIM * 2;
  unsigned short* zh = (unsigned short*)(ws + off); off += (size_t)BATCH * DIM * 2;
  unsigned short* zl = (unsigned short*)(ws + off); off += (size_t)BATCH * DIM * 2;
  const int pack = (ws_size >= off) ? 1 : 0;

  prep_p<<<1024, 256, 0, stream>>>(pr, psq, loss, ph, pl, pack);
  prep_z<<<BATCH, 256, 0, stream>>>(z, zcopy, zsq, zh, zl, pack);
  if (pack) {
    proto_gemm_async<<<dim3(NCLS / TN + 1, BATCH / TM), 256, 2 * BUF_USH * 2,
                       stream>>>(zh, zl, ph, pl, zsq, psq, scores);
  } else {
    proto_gemm_cvt<<<dim3(NCLS / TN + 1, BATCH / TM), 256, 0, stream>>>(
        z, pr, zsq, psq, scores);
  }
  proto_loss<<<BATCH, 256, 0, stream>>>(scores, tg, loss);
}
